// MultiHeadAttention_48232482734903
// MI455X (gfx1250) — matrix operand order repaired and checked by reading, not compiled, not run on hardware
//
#include <hip/hip_runtime.h>
#include <hip/hip_bf16.h>

typedef _Float16 v16h __attribute__((ext_vector_type(16)));
typedef _Float16 v8h  __attribute__((ext_vector_type(8)));
typedef float    v8f  __attribute__((ext_vector_type(8)));

#define B_  2
#define S_  2048
#define DM_ 1024
#define H_  16
#define HD_ 64

#define USE_ASYNC_LDS 1

template<typename T> struct is_h { static constexpr bool v = false; };
template<> struct is_h<_Float16> { static constexpr bool v = true; };

// ---------------------------------------------------------------------------
// Register-staged tile path (for f32 sources that need cvt to f16, and the
// K-major TRANSB path). fetch_* = global->regs, store_* = regs->LDS (as f16).
// ---------------------------------------------------------------------------
template<typename T, int ROWS, int BK, int NT>
__device__ inline void fetch_tile(const T* __restrict__ src, int ld, int tid,
                                  T* __restrict__ regs, bool pf) {
  constexpr int EPT = ROWS * BK / NT;
  const int idx = tid * EPT;
  const int r = idx / BK, c = idx % BK;
  const T* p = src + (size_t)r * ld + c;
#pragma unroll
  for (int i = 0; i < EPT; i++) regs[i] = p[i];
  if (pf) __builtin_prefetch(p + 2 * BK, 0, 1);   // two K-tiles ahead
}

template<typename T, int ROWS, int BK, int NT>
__device__ inline void store_tile(const T* __restrict__ regs,
                                  _Float16* __restrict__ dst, int tid) {
  constexpr int EPT = ROWS * BK / NT;
  const int idx = tid * EPT;
  const int r = idx / BK, c = idx % BK;
#pragma unroll
  for (int i = 0; i < EPT; i += 8) {
    v8h t;
#pragma unroll
    for (int j = 0; j < 8; j++) t[j] = (_Float16)regs[i + j];
    *(v8h*)&dst[r * BK + c + i] = t;
  }
}

template<int BN, int BK, int NT>
__device__ inline void fetch_tileBT(const _Float16* __restrict__ src, int ld,
                                    int tid, _Float16* __restrict__ regs) {
  constexpr int EPT = BN * BK / NT;            // 8 for BN=64
  static_assert(EPT == 8 && (NT / BN) * EPT == BK, "BT loader shape");
  const int n  = tid % BN;
  const int kk = (tid / BN) * EPT;
#pragma unroll
  for (int i = 0; i < EPT; i++) regs[i] = src[(size_t)(kk + i) * ld + n];
}

template<int BN, int BK, int NT>
__device__ inline void store_tileBT(const _Float16* __restrict__ regs,
                                    _Float16* __restrict__ dst, int tid) {
  const int n  = tid % BN;
  const int kk = (tid / BN) * 8;
  v8h t;
#pragma unroll
  for (int i = 0; i < 8; i++) t[i] = regs[i];
  *(v8h*)&dst[n * BK + kk] = t;
}

// ---------------------------------------------------------------------------
// Async global->LDS tile copy for f16 row-major tiles (no conversion needed):
// ROWS rows x 64B. Uses GLOBAL_LOAD_ASYNC_TO_LDS_B128 (ASYNCcnt), 64-bit
// vaddr + `off` form. LDS address = low 32 bits of the flat shared pointer.
// ---------------------------------------------------------------------------
#if USE_ASYNC_LDS
template<int ROWS, int NT>
__device__ inline void async_tile(const _Float16* __restrict__ src, int ld,
                                  _Float16* __restrict__ dst, int tid) {
  constexpr int CHUNKS = ROWS * 4;             // 16B chunks per 64B row
  constexpr int ITER = CHUNKS / NT;
  static_assert(ITER * NT == CHUNKS, "async tile shape");
#pragma unroll
  for (int it = 0; it < ITER; it++) {
    const int chunk = tid + it * NT;
    const int row = chunk >> 2, c16 = chunk & 3;
    const unsigned long long gaddr =
        (unsigned long long)src + (unsigned long long)(unsigned)(row * ld * 2 + c16 * 16);
    const unsigned laddr = (unsigned)(unsigned long long)(const void*)&dst[row * 32 + c16 * 8];
    asm volatile("global_load_async_to_lds_b128 %0, %1, off"
                 :: "v"(laddr), "v"(gaddr) : "memory");
  }
}
__device__ inline void wait_async() {
  asm volatile("s_wait_asynccnt 0x0" ::: "memory");
}
#endif

// ---------------------------------------------------------------------------
// Generic WMMA GEMM: C[m,n] = scale * sum_k A[m,k]*B[n,k] + bias[n]
// (TRANSB: B stored [K,N]). 256 threads = 8 waves, LDS double-buffered.
// ---------------------------------------------------------------------------
template<int BM, int BN, int WM, int WN, typename TA, typename TB, bool TRANSB>
__global__ __launch_bounds__(256)
void gemm_kernel(const TA* __restrict__ A, int lda, long sAb, long sAh,
                 const TB* __restrict__ Bm, int ldb, long sBb, long sBh,
                 float* __restrict__ outF, long sFb, long sFh,
                 _Float16* __restrict__ outH, long sHb, long sHh,
                 int ldc, const float* __restrict__ bias, float scale, int K) {
  constexpr int BK = 32, NT = 256;
  constexpr int WROWS = BM / WM, WCOLS = BN / WN;
  constexpr int TM = WROWS / 16, TN = WCOLS / 16;
#if USE_ASYNC_LDS
  constexpr bool AASYNC = is_h<TA>::v;               // f16 row-major A
  constexpr bool BASYNC = is_h<TB>::v && !TRANSB;    // f16 row-major B
#else
  constexpr bool AASYNC = false, BASYNC = false;
#endif

  const int z  = blockIdx.z;
  const long zb = z / H_, zh = z % H_;
  A  += zb * sAb + zh * sAh;
  Bm += zb * sBb + zh * sBh;
  if (outF) outF += zb * sFb + zh * sFh;
  if (outH) outH += zb * sHb + zh * sHh;

  const int blockM = blockIdx.y * BM;
  const int blockN = blockIdx.x * BN;

  __shared__ __align__(16) _Float16 As[2 * BM * BK];
  __shared__ __align__(16) _Float16 Bs[2 * BN * BK];

  const int tid  = threadIdx.x;
  const int wave = tid >> 5, lane = tid & 31;
  const int wm = wave / WN, wn = wave % WN;
  const int half = lane >> 4, l16 = lane & 15;

  const TA* Abase = A + (size_t)blockM * lda;
  const TB* Bbase;
  if constexpr (TRANSB) Bbase = Bm + blockN;
  else                  Bbase = Bm + (size_t)blockN * ldb;

  TA regA[BM * BK / NT];
  TB regB[BN * BK / NT];
  _Float16 regBT[8];

  v8f acc[TM][TN];
#pragma unroll
  for (int i = 0; i < TM; i++)
#pragma unroll
    for (int j = 0; j < TN; j++)
#pragma unroll
      for (int r = 0; r < 8; r++) acc[i][j][r] = 0.0f;

  // ---- prologue: tile 0 into buffer 0 ----
  if constexpr (AASYNC) {
    async_tile<BM, NT>((const _Float16*)Abase, lda, As, tid);
  } else {
    fetch_tile<TA, BM, BK, NT>(Abase, lda, tid, regA, BK < K);
    store_tile<TA, BM, BK, NT>(regA, As, tid);
  }
  if constexpr (BASYNC) {
    async_tile<BN, NT>((const _Float16*)Bbase, ldb, Bs, tid);
  } else if constexpr (TRANSB) {
    fetch_tileBT<BN, BK, NT>(Bbase, ldb, tid, regBT);
    store_tileBT<BN, BK, NT>(regBT, Bs, tid);
  } else {
    fetch_tile<TB, BN, BK, NT>(Bbase, ldb, tid, regB, BK < K);
    store_tile<TB, BN, BK, NT>(regB, Bs, tid);
  }
#if USE_ASYNC_LDS
  if constexpr (AASYNC || BASYNC) wait_async();
#endif
  __syncthreads();

  int cur = 0;
  for (int k0 = 0; k0 < K; k0 += BK) {
    const bool more = (k0 + BK) < K;
    const int nxt = cur ^ 1;

    // ---- kick off next tile (async straight to LDS, f32 into regs) ----
    if (more) {
      if constexpr (AASYNC)
        async_tile<BM, NT>((const _Float16*)Abase + k0 + BK, lda, As + nxt * BM * BK, tid);
      else
        fetch_tile<TA, BM, BK, NT>(Abase + k0 + BK, lda, tid, regA, k0 + 2 * BK < K);
      if constexpr (BASYNC)
        async_tile<BN, NT>((const _Float16*)Bbase + k0 + BK, ldb, Bs + nxt * BN * BK, tid);
      else if constexpr (TRANSB)
        fetch_tileBT<BN, BK, NT>(Bbase + (size_t)(k0 + BK) * ldb, ldb, tid, regBT);
      else
        fetch_tile<TB, BN, BK, NT>(Bbase + k0 + BK, ldb, tid, regB, k0 + 2 * BK < K);
    }

    // ---- compute from buffer `cur` ----
    const _Float16* Asc = As + cur * BM * BK;
    const _Float16* Bsc = Bs + cur * BN * BK;

    // B fragments: 32x16, lane = column (l16), lanes>=16 hold K=16..31.
    v16h bfrag[TN];
#pragma unroll
    for (int tn = 0; tn < TN; tn++) {
      const int col = wn * WCOLS + tn * 16 + l16;
      const v8h* pb = (const v8h*)&Bsc[col * BK + half * 16];
      v8h b0 = pb[0], b1 = pb[1];
#pragma unroll
      for (int i = 0; i < 8; i++) { bfrag[tn][i] = b0[i]; bfrag[tn][8 + i] = b1[i]; }
    }
#pragma unroll
    for (int tm = 0; tm < TM; tm++) {
      // A fragment: 16x32; lanes<16: K 0..15, lanes>=16: K 16..31.
      const int row = wm * WROWS + tm * 16 + l16;
      v8h a0 = *(const v8h*)&Asc[row * BK + half * 16];
      v8h a1 = *(const v8h*)&Asc[row * BK + half * 16 + 8];
      v16h afrag;
#pragma unroll
      for (int i = 0; i < 8; i++) { afrag[i] = a0[i]; afrag[8 + i] = a1[i]; }
#pragma unroll
      for (int tn = 0; tn < TN; tn++)
        acc[tm][tn] = __builtin_amdgcn_wmma_f32_16x16x32_f16(
            false, afrag, false, bfrag[tn], (short)0, acc[tm][tn], false, false);
    }

    // ---- commit next tile and swap ----
    if (more) {
      if constexpr (!AASYNC)
        store_tile<TA, BM, BK, NT>(regA, As + nxt * BM * BK, tid);
      if constexpr (!BASYNC) {
        if constexpr (TRANSB) store_tileBT<BN, BK, NT>(regBT, Bs + nxt * BN * BK, tid);
        else                  store_tile<TB, BN, BK, NT>(regB, Bs + nxt * BN * BK, tid);
      }
#if USE_ASYNC_LDS
      if constexpr (AASYNC || BASYNC) wait_async();
#endif
      __syncthreads();
      cur = nxt;
    }
  }

  // ---- epilogue: D layout — VGPR r holds M = r + 8*(lane/16), N = lane%16 ----
#pragma unroll
  for (int tm = 0; tm < TM; tm++)
#pragma unroll
    for (int tn = 0; tn < TN; tn++) {
      const int col = blockN + wn * WCOLS + tn * 16 + l16;
      const float bv = bias ? bias[col] : 0.0f;
#pragma unroll
      for (int r = 0; r < 8; r++) {
        const int row = blockM + wm * WROWS + tm * 16 + half * 8 + r;
        const float val = acc[tm][tn][r] * scale + bv;
        if (outF) __builtin_nontemporal_store(val, &outF[(size_t)row * ldc + col]);
        if (outH) outH[(size_t)row * ldc + col] = (_Float16)val;
      }
    }
}

// ---------------------------------------------------------------------------
// Masked softmax, one block (256 thr, 8 waves) per attention row of 2048.
// ---------------------------------------------------------------------------
__global__ __launch_bounds__(256)
void softmax_kernel(float* __restrict__ attn, const int* __restrict__ mask) {
  const long row = blockIdx.x;            // over B*H*S
  const int q  = (int)(row % S_);
  const int bh = (int)(row / S_);
  const int b  = bh / H_;
  float* p = attn + row * (long)S_;
  const int* mrow = mask + ((long)b * S_ + q) * S_;

  const int tid = threadIdx.x, lane = tid & 31, wave = tid >> 5;
  __shared__ float red[8];

  float vals[8];
  float mx = -__builtin_inff();
#pragma unroll
  for (int j = 0; j < 8; j++) {
    const int i = tid + j * 256;
    float v = (mrow[i] == 0) ? -__builtin_inff() : p[i];
    vals[j] = v;
    mx = fmaxf(mx, v);
  }
#pragma unroll
  for (int o = 16; o > 0; o >>= 1) mx = fmaxf(mx, __shfl_xor(mx, o, 32));
  if (lane == 0) red[wave] = mx;
  __syncthreads();
  mx = red[0];
#pragma unroll
  for (int w = 1; w < 8; w++) mx = fmaxf(mx, red[w]);
  __syncthreads();

  float sm = 0.0f;
#pragma unroll
  for (int j = 0; j < 8; j++) {
    const float e = __expf(vals[j] - mx);
    vals[j] = e;
    sm += e;
  }
#pragma unroll
  for (int o = 16; o > 0; o >>= 1) sm += __shfl_xor(sm, o, 32);
  if (lane == 0) red[wave] = sm;
  __syncthreads();
  sm = red[0];
#pragma unroll
  for (int w = 1; w < 8; w++) sm += red[w];
  const float inv = 1.0f / sm;
#pragma unroll
  for (int j = 0; j < 8; j++)
    __builtin_nontemporal_store(vals[j] * inv, &p[tid + j * 256]);
}

// ---------------------------------------------------------------------------
extern "C" void kernel_launch(void* const* d_in, const int* in_sizes, int n_in,
                              void* d_out, int out_size, void* d_ws, size_t ws_size,
                              hipStream_t stream) {
  const float* q  = (const float*)d_in[0];
  const float* k  = (const float*)d_in[1];
  const float* v  = (const float*)d_in[2];
  const int* mask = (const int*)d_in[3];
  const float* Wq = (const float*)d_in[4];  const float* bq = (const float*)d_in[5];
  const float* Wk = (const float*)d_in[6];  const float* bk = (const float*)d_in[7];
  const float* Wv = (const float*)d_in[8];  const float* bv = (const float*)d_in[9];
  const float* Wf = (const float*)d_in[10]; const float* bf = (const float*)d_in[11];

  float* out  = (float*)d_out;                          // [B,S,DM]
  float* attn = out + (size_t)B_ * S_ * DM_;            // [B,H,S,S]

  const size_t NTOK = (size_t)B_ * S_;                  // 4096
  _Float16* qh  = (_Float16*)d_ws;                      // [B*S, DM] f16
  _Float16* kh  = qh  + NTOK * DM_;
  _Float16* vh  = kh  + NTOK * DM_;
  _Float16* ctx = vh  + NTOK * DM_;

  const dim3 blk(256);

  // 1) Q/K/V projections: x @ W^T + b  (M=4096, N=1024, K=1024) -> f16 ws
  {
    const dim3 g(DM_ / 128, (unsigned)(NTOK / 128), 1);
    gemm_kernel<128,128,2,4, float, float, false><<<g, blk, 0, stream>>>(
        q, DM_, 0, 0, Wq, DM_, 0, 0, nullptr, 0, 0, qh, 0, 0, DM_, bq, 1.0f, DM_);
    gemm_kernel<128,128,2,4, float, float, false><<<g, blk, 0, stream>>>(
        k, DM_, 0, 0, Wk, DM_, 0, 0, nullptr, 0, 0, kh, 0, 0, DM_, bk, 1.0f, DM_);
    gemm_kernel<128,128,2,4, float, float, false><<<g, blk, 0, stream>>>(
        v, DM_, 0, 0, Wv, DM_, 0, 0, nullptr, 0, 0, vh, 0, 0, DM_, bv, 1.0f, DM_);
  }

  // 2) Scores per (b,h): qh @ kh^T / 8  (M=N=2048, K=64) -> attn region f32
  {
    const dim3 g(S_ / 128, S_ / 128, B_ * H_);
    gemm_kernel<128,128,2,4, _Float16, _Float16, false><<<g, blk, 0, stream>>>(
        qh, DM_, (long)S_ * DM_, HD_,
        kh, DM_, (long)S_ * DM_, HD_,
        attn, (long)H_ * S_ * S_, (long)S_ * S_,
        nullptr, 0, 0, S_, nullptr, 0.125f, HD_);
  }

  // 3) Softmax in place on attn (also the 2nd reference output)
  softmax_kernel<<<dim3((unsigned)(B_ * H_ * S_)), blk, 0, stream>>>(attn, mask);

  // 4) ctx = attn @ vh per (b,h)  (M=2048, N=64, K=2048); vh is K-major [k,d]
  {
    const dim3 g(HD_ / 64, S_ / 128, B_ * H_);
    gemm_kernel<128,64,4,2, float, _Float16, true><<<g, blk, 0, stream>>>(
        attn, S_, (long)H_ * S_ * S_, (long)S_ * S_,
        vh, DM_, (long)S_ * DM_, HD_,
        nullptr, 0, 0,
        ctx, (long)S_ * DM_, HD_, DM_, nullptr, 1.0f, S_);
  }

  // 5) out = ctx @ Wf^T + bf  (M=4096, N=1024, K=1024) -> f32 d_out
  {
    const dim3 g(DM_ / 128, (unsigned)(NTOK / 128), 1);
    gemm_kernel<128,128,2,4, _Float16, float, false><<<g, blk, 0, stream>>>(
        ctx, DM_, 0, 0, Wf, DM_, 0, 0, out, 0, 0, nullptr, 0, 0, DM_, bf, 1.0f, DM_);
  }
}
